// NeuralECMTokenModel_15307263443322
// MI455X (gfx1250) — matrix operand order repaired and checked by reading, compiled, not run on hardware
//
#include <hip/hip_runtime.h>

// ---------------------------------------------------------------------------
// GAT layer for MI455X (gfx1250, wave32, WMMA).
//   - Dominant GEMM: v_wmma_f32_16x16x32_bf16 (bf16 in, f32 accumulate).
//   - W held as B-fragments in VGPRs (node-invariant, 128 VGPRs/wave):
//     inner loop is only A ds_loads + WMMA. LDS ~54KB -> 2 workgroups/WGP.
//   - neighbors (the 655MB HBM stream) read exactly once, software-pipelined:
//     next node's 32KB prefetched into registers during current node's GEMM.
// ---------------------------------------------------------------------------

typedef __attribute__((ext_vector_type(16))) __bf16        v16bf;
typedef __attribute__((ext_vector_type(8)))  float         v8f;
typedef __attribute__((ext_vector_type(4)))  unsigned int  v4u;
typedef __attribute__((ext_vector_type(2)))  unsigned int  v2u;
typedef __attribute__((ext_vector_type(4)))  float         v4f;

#define DFEAT 256
#define KNB   32

// ---- LDS layout (bytes), dynamic shared memory ----------------------------
#define OFF_NB    0        // 32 rows x NB_PITCH bf16 (padded): 16896
#define NB_PITCH  528      // 264 bf16/row -> bank spread, 16B-aligned rows
#define OFF_PROJ  16896    // 32 x 256 f32 = 32768
#define OFF_ASRC  49664    // 256 f32
#define OFF_VT    50688    // 256 f32
#define OFF_BIAS  51712    // 256 f32
#define OFF_RED   52736    // 256 f32 (scores_t partials)
#define OFF_SC    53760    // 32x8 f32 (scores_s partials)
#define OFF_EXP   54784    // 32 f32
#define OFF_MISC  54912    // 16 f32 scratch
#define SMEM_BYTES 54976

__device__ __forceinline__ unsigned short f2bf(float x) {
  unsigned int u = __float_as_uint(x);
  u += 0x7FFFu + ((u >> 16) & 1u);      // round-to-nearest-even
  return (unsigned short)(u >> 16);
}

union FragAB { v4u q[2]; unsigned int u[8]; v16bf v; };

// ---------------------------------------------------------------------------
// Kernel 1: v_t[f] = sum_o W[o,f] * a_tgt[o]   (collapses nodes@W.T@a_tgt)
// ---------------------------------------------------------------------------
__global__ void gat_vt_kernel(const float* __restrict__ W,
                              const float* __restrict__ a_tgt,
                              float* __restrict__ vt) {
  const int f = threadIdx.x;            // 256 threads
  float s = 0.f;
  for (int o = 0; o < DFEAT; ++o)
    s += W[o * DFEAT + f] * a_tgt[o];   // coalesced across threads
  vt[f] = s;
}

// ---------------------------------------------------------------------------
// Main kernel: 256 threads = 8 waves; persistent blocks over nodes.
// ---------------------------------------------------------------------------
extern "C" __global__ __launch_bounds__(256, 1)
void gat_main_kernel(const float* __restrict__ nodes,
                     const float* __restrict__ neighbors,
                     const float* __restrict__ W,
                     const float* __restrict__ a_src,
                     const float* __restrict__ bias,
                     const float* __restrict__ vt,
                     float* __restrict__ out,
                     int N) {
  extern __shared__ char smem[];
  const int t = threadIdx.x;
  const int w = t >> 5;                 // wave id 0..7
  const int l = t & 31;                 // lane id
  const int lrow = l & 15;
  const int half = l >> 4;

  const int nt0 = w * 2, nt1 = nt0 + 1;
  const int o0 = nt0 * 16 + lrow, o1 = nt1 * 16 + lrow;

  // ---- One-time per block: small vectors to LDS ---------------------------
  ((float*)(smem + OFF_ASRC))[t] = a_src[t];
  ((float*)(smem + OFF_VT))[t]   = vt[t];
  ((float*)(smem + OFF_BIAS))[t] = bias[t];

  // ---- One-time per wave: build node-invariant B fragments in VGPRs ------
  // B tile (nt, ks) is 32(K=f) x 16(N=o). Lane l: column o = nt*16 + l%16,
  // 16 values kk = (l/16)*16 + e, e ascending == 16 consecutive f of W row o.
  v16bf Bf[2][8];
  #pragma unroll
  for (int j = 0; j < 2; ++j) {
    const float* wrow = W + (size_t)((nt0 + j) * 16 + lrow) * DFEAT + half * 16;
    #pragma unroll
    for (int ks = 0; ks < 8; ++ks) {
      const v4f* p = (const v4f*)(wrow + ks * 32);
      FragAB fb;
      #pragma unroll
      for (int q = 0; q < 4; ++q) {
        v4f v = p[q];
        fb.u[q * 2 + 0] = (unsigned)f2bf(v.x) | ((unsigned)f2bf(v.y) << 16);
        fb.u[q * 2 + 1] = (unsigned)f2bf(v.z) | ((unsigned)f2bf(v.w) << 16);
      }
      Bf[j][ks] = fb.v;
    }
  }
  __syncthreads();

  // ---- Software-pipelined node loop --------------------------------------
  v4f pf[8];                            // prefetch buffer: 32KB/block of nbrs
  int n = blockIdx.x;
  if (n < N) {
    const v4f* nb4 = (const v4f*)(neighbors + (size_t)n * (KNB * DFEAT));
    #pragma unroll
    for (int i = 0; i < 8; ++i) pf[i] = nb4[i * 256 + t];
  }

  for (; n < N; n += gridDim.x) {
    // ---- Drain prefetch: f32 -> bf16 into padded LDS ---------------------
    #pragma unroll
    for (int i = 0; i < 8; ++i) {
      int g4 = i * 256 + t;             // 2048 float4s = 32x256 floats
      int row = g4 >> 6;
      int col = (g4 & 63) << 2;         // float column
      v2u pk;
      pk.x = (unsigned)f2bf(pf[i].x) | ((unsigned)f2bf(pf[i].y) << 16);
      pk.y = (unsigned)f2bf(pf[i].z) | ((unsigned)f2bf(pf[i].w) << 16);
      *(v2u*)(smem + OFF_NB + row * NB_PITCH + col * 2) = pk;
    }
    // ---- Issue next node's prefetch (overlaps GEMM + attention below) ----
    {
      int nn = n + gridDim.x;
      if (nn < N) {
        const v4f* nb4 = (const v4f*)(neighbors + (size_t)nn * (KNB * DFEAT));
        #pragma unroll
        for (int i = 0; i < 8; ++i) pf[i] = nb4[i * 256 + t];
      }
    }
    __syncthreads();

    // ---- GEMM: proj[k,o] = sum_f nb[k,f]*W[o,f]; A from LDS, B in VGPRs --
    v8f acc00 = {0.f,0.f,0.f,0.f,0.f,0.f,0.f,0.f};
    v8f acc01 = acc00, acc10 = acc00, acc11 = acc00;
    #pragma unroll
    for (int ks = 0; ks < 8; ++ks) {
      // A 16x32 bf16 layout: lane half selects K {0..15} or {16..31};
      // one contiguous 32B run in the LDS row (matches B's k order).
      FragAB A0, A1;
      const char* r0 = smem + OFF_NB + (lrow)      * NB_PITCH + (ks * 32 + half * 16) * 2;
      const char* r1 = smem + OFF_NB + (16 + lrow) * NB_PITCH + (ks * 32 + half * 16) * 2;
      A0.q[0] = *(const v4u*)(r0); A0.q[1] = *(const v4u*)(r0 + 16);
      A1.q[0] = *(const v4u*)(r1); A1.q[1] = *(const v4u*)(r1 + 16);
      acc00 = __builtin_amdgcn_wmma_f32_16x16x32_bf16(false, A0.v, false, Bf[0][ks], (short)0, acc00, false, false);
      acc01 = __builtin_amdgcn_wmma_f32_16x16x32_bf16(false, A0.v, false, Bf[1][ks], (short)0, acc01, false, false);
      acc10 = __builtin_amdgcn_wmma_f32_16x16x32_bf16(false, A1.v, false, Bf[0][ks], (short)0, acc10, false, false);
      acc11 = __builtin_amdgcn_wmma_f32_16x16x32_bf16(false, A1.v, false, Bf[1][ks], (short)0, acc11, false, false);
    }

    // ---- Spill proj tiles to LDS (C layout: M=(l/16)*8+r, N=l%16) --------
    {
      float* proj = (float*)(smem + OFF_PROJ);
      const int kb = half * 8;
      #pragma unroll
      for (int r = 0; r < 8; ++r) {
        proj[(kb + r)      * DFEAT + o0] = acc00[r];
        proj[(kb + r)      * DFEAT + o1] = acc01[r];
        proj[(16 + kb + r) * DFEAT + o0] = acc10[r];
        proj[(16 + kb + r) * DFEAT + o1] = acc11[r];
      }
    }
    __syncthreads();

    // ---- Attention: scores, softmax (raw exp + EPS), weighted sum, ELU ---
    {
      // scores_s partials: thread t -> k = t/8, o-segment = (t%8)*32
      const float* projk = (const float*)(smem + OFF_PROJ) + (t >> 3) * DFEAT;
      const float* asrc  = (const float*)(smem + OFF_ASRC);
      const int seg = (t & 7) * 32;
      float sp = 0.f;
      #pragma unroll
      for (int j = 0; j < 32; ++j) sp += projk[seg + j] * asrc[seg + j];
      ((float*)(smem + OFF_SC))[t] = sp;
      // scores_t partials: nodes[n,:] . v_t
      ((float*)(smem + OFF_RED))[t] =
          nodes[(size_t)n * DFEAT + t] * ((const float*)(smem + OFF_VT))[t];
    }
    __syncthreads();
    if (t < 8) {   // fixed-order two-level reduction (deterministic)
      const float* red = (const float*)(smem + OFF_RED);
      float s = 0.f;
      for (int j = 0; j < 32; ++j) s += red[t * 32 + j];
      ((float*)(smem + OFF_MISC))[2 + t] = s;
    }
    __syncthreads();
    if (t == 0) {
      const float* m = (const float*)(smem + OFF_MISC);
      float st = 0.f;
      for (int j = 0; j < 8; ++j) st += m[2 + j];
      ((float*)(smem + OFF_MISC))[0] = st;
    }
    __syncthreads();
    if (t < 32) {
      const float* sc = (const float*)(smem + OFF_SC);
      float e = 0.f;
      for (int j = 0; j < 8; ++j) e += sc[t * 8 + j];
      e += ((const float*)(smem + OFF_MISC))[0];
      e = (e >= 0.f) ? e : 0.2f * e;                 // leakyReLU(0.2)
      ((float*)(smem + OFF_EXP))[t] = expf(e);       // raw exp (as reference)
    }
    __syncthreads();
    if (t == 0) {
      const float* ev = (const float*)(smem + OFF_EXP);
      float d = 0.f;
      for (int j = 0; j < 32; ++j) d += ev[j];
      ((float*)(smem + OFF_MISC))[1] = 1.0f / (d + 1e-16f);
    }
    __syncthreads();
    {
      const float* ev   = (const float*)(smem + OFF_EXP);
      const float* proj = (const float*)(smem + OFF_PROJ);
      float acc = 0.f;
      #pragma unroll
      for (int k = 0; k < KNB; ++k) acc += ev[k] * proj[k * DFEAT + t];
      acc = acc * ((const float*)(smem + OFF_MISC))[1]
            + ((const float*)(smem + OFF_BIAS))[t];
      out[(size_t)n * DFEAT + t] = (acc > 0.f) ? acc : expm1f(acc);  // ELU
    }
    __syncthreads();   // protect LDS reuse for next node
  }
}

// ---------------------------------------------------------------------------
extern "C" void kernel_launch(void* const* d_in, const int* in_sizes, int n_in,
                              void* d_out, int out_size, void* d_ws, size_t ws_size,
                              hipStream_t stream) {
  const float* nodes     = (const float*)d_in[0];
  const float* neighbors = (const float*)d_in[1];
  const float* W         = (const float*)d_in[2];
  const float* a_src     = (const float*)d_in[3];
  const float* a_tgt     = (const float*)d_in[4];
  const float* bias      = (const float*)d_in[5];
  float* out = (float*)d_out;
  float* vt  = (float*)d_ws;          // 256 floats of scratch

  const int N = in_sizes[0] / DFEAT;  // 20000

  gat_vt_kernel<<<1, 256, 0, stream>>>(W, a_tgt, vt);

  int grid = 1024;                    // persistent blocks (~20 nodes each);
  if (grid > N) grid = N;             // amortizes the per-wave W frag build
  gat_main_kernel<<<grid, 256, SMEM_BYTES, stream>>>(
      nodes, neighbors, W, a_src, bias, vt, out, N);
}
